// BinaryConv2d_78108275245419
// MI455X (gfx1250) — 8-bit matrix operand order repaired and checked by reading, compiled, not run on hardware
//
#include <hip/hip_runtime.h>

typedef int v8i __attribute__((ext_vector_type(8)));

#define HH 56
#define WW 56
#define CIN 256
#define COUT 256
#define HWX (HH * WW)
#define KTOT (CIN * 9)
#define M_TILE 128
#define NCHUNK (KTOT / 64)

#define A_ROW 80
#define B_ROW 80
#define A_BYTES (128 * A_ROW)
#define B_BYTES (256 * B_ROW)
#define OFF_A0 0
#define OFF_A1 (A_BYTES)
#define OFF_B0 (2 * A_BYTES)
#define OFF_B1 (2 * A_BYTES + B_BYTES)
#define SMEM_BYTES (2 * A_BYTES + 2 * B_BYTES)

__device__ __forceinline__ void async_copy_b128(unsigned ldsOff, unsigned gOff,
                                                const signed char* base) {
    asm volatile("global_load_async_to_lds_b128 %0, %1, %2"
                 :: "v"(ldsOff), "v"(gOff), "s"(base) : "memory");
}
__device__ __forceinline__ void wait_async0() {
    asm volatile("s_wait_asynccnt 0x0" ::: "memory");
}

__global__ __launch_bounds__(256) void binx_kernel(const float* __restrict__ x,
                                                   signed char* __restrict__ bx) {
    __shared__ __align__(4) char tile[56 * 264];
    const int n = blockIdx.x / HH;
    const int h = blockIdx.x % HH;
    const int tid = threadIdx.x;
    const float* xp = x + (size_t)n * CIN * HWX + (size_t)h * WW;
#pragma unroll 4
    for (int it = 0; it < 56; ++it) {
        int idx = it * 256 + tid;
        int c = idx / 56;
        int w = idx % 56;
        float v = xp[(size_t)c * HWX + w];
        tile[w * 264 + c] = (v > 0.f) ? (signed char)1 : ((v < 0.f) ? (signed char)-1 : (signed char)0);
    }
    __syncthreads();
    int* dst = (int*)(bx + (size_t)(n * HH + h) * WW * CIN);
#pragma unroll
    for (int it = 0; it < 14; ++it) {
        int idx = it * 256 + tid;
        int w = idx >> 6;
        int d = idx & 63;
        dst[idx] = *(const int*)(tile + w * 264 + d * 4);
    }
}

__global__ __launch_bounds__(256) void binw_kernel(const float* __restrict__ wt,
                                                   signed char* __restrict__ bw) {
    int gid = blockIdx.x * 256 + threadIdx.x;
    int r4 = gid * 4;
    int o = r4 / KTOT;
    int r = r4 % KTOT;
    int tap = r >> 8;
    int c = r & 255;
    const float* wp = wt + (size_t)o * KTOT + (size_t)c * 9 + tap;
    unsigned int pk = 0;
#pragma unroll
    for (int e = 0; e < 4; ++e) {
        float v = wp[e * 9];
        unsigned int s = (v > 0.f) ? 1u : ((v < 0.f) ? 0xFFu : 0u);
        pk |= s << (8 * e);
    }
    ((unsigned int*)bw)[gid] = pk;
}

__global__ __launch_bounds__(256) void bconv_wmma_kernel(const signed char* __restrict__ bx,
                                                         const signed char* __restrict__ bw,
                                                         const float* __restrict__ alpha,
                                                         float* __restrict__ out) {
    __shared__ __align__(16) char smem[SMEM_BYTES];
    const unsigned smemBase = (unsigned)(size_t)(void*)&smem[0];

    const int tid = threadIdx.x;
    const int lane = tid & 31;
    const int wv = tid >> 5;
    const int g = lane >> 4;
    const int ml = lane & 15;

    const int mBase = (wv >> 2) * 64;
    const int nlb = (wv & 3) * 64;

    const int tileM = blockIdx.x * M_TILE;

    const int arow0 = tid >> 2;
    const int arow1 = arow0 + 64;
    const int aseg16 = (tid & 3) * 16;
    const int p0 = tileM + arow0;
    const int p1 = tileM + arow1;
    const int n0 = p0 / HWX, rem0 = p0 % HWX;
    const int n1 = p1 / HWX, rem1 = p1 % HWX;
    const int aho0 = rem0 / WW, awo0 = rem0 % WW;
    const int aho1 = rem1 / WW, awo1 = rem1 % WW;

    v8i acc[4][4];
#pragma unroll
    for (int i = 0; i < 4; ++i)
#pragma unroll
        for (int j = 0; j < 4; ++j) acc[i][j] = (v8i){0, 0, 0, 0, 0, 0, 0, 0};

    auto issueChunk = [&](int kc, unsigned aOff, unsigned bOff) {
        const int tap = kc >> 2;
        const int c0 = (kc & 3) * 64;
        const int kh = tap / 3, kw = tap % 3;
        {
            const int hi = aho0 + kh - 1, wi = awo0 + kw - 1;
            if ((unsigned)hi < HH && (unsigned)wi < WW) {
                unsigned gOff = (unsigned)(((n0 * HH + hi) * WW + wi) * CIN + c0 + aseg16);
                async_copy_b128(smemBase + aOff + (unsigned)(arow0 * A_ROW + aseg16), gOff, bx);
            } else {
                *(int4*)(smem + aOff + arow0 * A_ROW + aseg16) = make_int4(0, 0, 0, 0);
            }
        }
        {
            const int hi = aho1 + kh - 1, wi = awo1 + kw - 1;
            if ((unsigned)hi < HH && (unsigned)wi < WW) {
                unsigned gOff = (unsigned)(((n1 * HH + hi) * WW + wi) * CIN + c0 + aseg16);
                async_copy_b128(smemBase + aOff + (unsigned)(arow1 * A_ROW + aseg16), gOff, bx);
            } else {
                *(int4*)(smem + aOff + arow1 * A_ROW + aseg16) = make_int4(0, 0, 0, 0);
            }
        }
        const unsigned kOff = (unsigned)(tap * CIN + c0);
#pragma unroll
        for (int s = 0; s < 4; ++s) {
            int sidx = tid + 256 * s;
            int row = sidx >> 2;
            int seg = (sidx & 3) * 16;
            async_copy_b128(smemBase + bOff + (unsigned)(row * B_ROW + seg),
                            (unsigned)(row * KTOT) + kOff + (unsigned)seg, bw);
        }
    };

    auto loadA = [&](const char* Ac, int r16) -> v8i {
        const char* ab = Ac + (r16 + ml) * A_ROW;
        int2 a0 = *(const int2*)(ab + g * 8);
        int2 a1 = *(const int2*)(ab + 16 + g * 8);
        int2 a2 = *(const int2*)(ab + 32 + g * 8);
        int2 a3 = *(const int2*)(ab + 48 + g * 8);
        v8i A;
        A[0] = a0.x; A[1] = a0.y; A[2] = a1.x; A[3] = a1.y;
        A[4] = a2.x; A[5] = a2.y; A[6] = a3.x; A[7] = a3.y;
        return A;
    };
    auto loadB = [&](const char* Bc, int c16) -> v8i {
        const char* bb = Bc + (c16 + ml) * B_ROW;
        int2 b0 = *(const int2*)(bb + g * 8);
        int2 b1 = *(const int2*)(bb + 16 + g * 8);
        int2 b2 = *(const int2*)(bb + 32 + g * 8);
        int2 b3 = *(const int2*)(bb + 48 + g * 8);
        v8i B;
        B[0] = b0.x; B[1] = b0.y; B[2] = b1.x; B[3] = b1.y;
        B[4] = b2.x; B[5] = b2.y; B[6] = b3.x; B[7] = b3.y;
        return B;
    };

    issueChunk(0, OFF_A0, OFF_B0);
    wait_async0();
    __syncthreads();

    for (int kc = 0; kc < NCHUNK; ++kc) {
        const char* Ac = smem + ((kc & 1) ? OFF_A1 : OFF_A0);
        const char* Bc = smem + ((kc & 1) ? OFF_B1 : OFF_B0);
        if (kc + 1 < NCHUNK)
            issueChunk(kc + 1, (kc & 1) ? OFF_A0 : OFF_A1, (kc & 1) ? OFF_B0 : OFF_B1);

        v8i Af0 = loadA(Ac, mBase);
        v8i Af1 = loadA(Ac, mBase + 16);
        v8i Af2 = loadA(Ac, mBase + 32);
        v8i Af3 = loadA(Ac, mBase + 48);
#pragma unroll
        for (int t = 0; t < 4; ++t) {
            v8i Bf = loadB(Bc, nlb + t * 16);
            acc[0][t] = __builtin_amdgcn_wmma_i32_16x16x64_iu8(true, Af0, true, Bf,
                                                               acc[0][t], false, false);
            acc[1][t] = __builtin_amdgcn_wmma_i32_16x16x64_iu8(true, Af1, true, Bf,
                                                               acc[1][t], false, false);
            acc[2][t] = __builtin_amdgcn_wmma_i32_16x16x64_iu8(true, Af2, true, Bf,
                                                               acc[2][t], false, false);
            acc[3][t] = __builtin_amdgcn_wmma_i32_16x16x64_iu8(true, Af3, true, Bf,
                                                               acc[3][t], false, false);
        }

        if (kc + 1 < NCHUNK) wait_async0();
        __syncthreads();
    }

    int* const csm = (int*)smem;
    const float aV = alpha[0];
    const int u = tid >> 1;
    const int ch = u & 63;
    const int pixBase = ((u >> 6) << 6) + (tid & 1) * 32;
    const int pAbs = tileM + pixBase;
    const int nImg = pAbs / HWX;
    const int prem = pAbs % HWX;
#pragma unroll
    for (int h = 0; h < 4; ++h) {
        if ((wv & 3) == h) {
#pragma unroll
            for (int ms = 0; ms < 4; ++ms)
#pragma unroll
                for (int t = 0; t < 4; ++t)
#pragma unroll
                    for (int r = 0; r < 8; ++r) {
                        const int b0 = (mBase + ms * 16 + r + g * 8) * 68 + t * 16 + ml;
                        csm[b0] = acc[ms][t][r];
                    }
        }
        __syncthreads();
        float* op = out + ((size_t)(nImg * COUT + h * 64 + ch) * HWX + prem);
#pragma unroll
        for (int q = 0; q < 8; ++q) {
            const int m0 = pixBase + q * 4;
            float4 o;
            o.x = aV * (float)csm[(m0 + 0) * 68 + ch];
            o.y = aV * (float)csm[(m0 + 1) * 68 + ch];
            o.z = aV * (float)csm[(m0 + 2) * 68 + ch];
            o.w = aV * (float)csm[(m0 + 3) * 68 + ch];
            *(float4*)(op + q * 4) = o;
        }
        __syncthreads();
    }
}

extern "C" void kernel_launch(void* const* d_in, const int* in_sizes, int n_in,
                              void* d_out, int out_size, void* d_ws, size_t ws_size,
                              hipStream_t stream) {
    const float* x = (const float*)d_in[0];
    const float* wt = (const float*)d_in[1];
    const float* alpha = (const float*)d_in[2];
    float* out = (float*)d_out;

    signed char* bx = (signed char*)d_ws;
    signed char* bw = bx + (size_t)32 * HH * WW * CIN;

    binx_kernel<<<32 * HH, 256, 0, stream>>>(x, bx);
    binw_kernel<<<(COUT * KTOT) / (4 * 256), 256, 0, stream>>>(wt, bw);

    bconv_wmma_kernel<<<dim3((32 * HWX) / M_TILE), 256, 0, stream>>>(bx, bw, alpha, out);
}
